// AttentionSubstructurePooling_13597866459657
// MI455X (gfx1250) — matrix operand order repaired and checked by reading, not compiled, not run on hardware
//
#include <hip/hip_runtime.h>
#include <cstdint>

// ---------------- constants ----------------
static constexpr int kE  = 512;     // embedding
static constexpr int kH  = 8;       // heads
static constexpr int kN  = 131072;  // nodes
static constexpr int kB  = 4096;    // graphs

typedef __attribute__((ext_vector_type(16))) __bf16 v16bf;
typedef __attribute__((ext_vector_type(8)))  __bf16 v8bf_t;
typedef __attribute__((ext_vector_type(8)))  float  v8f;
typedef __attribute__((ext_vector_type(4)))  float  f4v;

static __device__ __forceinline__ int lowb(const int* __restrict__ a, int n, int v) {
  int lo = 0, hi = n;
  while (lo < hi) { int mid = (lo + hi) >> 1; if (a[mid] < v) lo = mid + 1; else hi = mid; }
  return lo;
}

// ---------------------------------------------------------------------------
// fp32 -> bf16 conversion (native v_cvt, 2 elements per thread)
// ---------------------------------------------------------------------------
__global__ __launch_bounds__(256)
void f32_to_bf16_kernel(const float* __restrict__ in, __bf16* __restrict__ out, int n)
{
  const int i = (blockIdx.x * 256 + threadIdx.x) * 2;
  if (i < n) {
    out[i]     = (__bf16)in[i];
    out[i + 1] = (__bf16)in[i + 1];
  }
}

// ---------------------------------------------------------------------------
// fp32 -> bf16 transposed copy: out[j, i] = in[i, j]   (rows x cols, LDS tile)
// ---------------------------------------------------------------------------
__global__ __launch_bounds__(256)
void transpose_f32_to_bf16(const float* __restrict__ in, __bf16* __restrict__ out,
                           int rows, int cols)
{
  __shared__ float t[32][33];
  const int bx = blockIdx.x * 32;     // col tile
  const int by = blockIdx.y * 32;     // row tile
  const int lx = threadIdx.x & 31;
  const int ly = threadIdx.x >> 5;    // 8 sub-rows
  #pragma unroll
  for (int r = ly; r < 32; r += 8)
    t[r][lx] = in[(size_t)(by + r) * cols + bx + lx];
  __syncthreads();
  #pragma unroll
  for (int r = ly; r < 32; r += 8)
    out[(size_t)(bx + r) * rows + by + lx] = (__bf16)t[lx][r];
}

// ---------------------------------------------------------------------------
// bf16 WMMA GEMM:  D[M,N] = A[M,K] x B[N,K]^T (+bias), fp32 accumulate.
// Outputs (each optional): C fp32 [M,N]; Cbf bf16 [M,N]; CbfT bf16 [N,M].
// 256 threads = 8 waves; block tile 128(M) x 64(N); wave tile 32x32 = 2x2 WMMA.
// Requires M%128==0, N%64==0, K%32==0 (true for all shapes used here).
// ---------------------------------------------------------------------------
__global__ __launch_bounds__(256)
void gemm_bf16_wmma(const __bf16* __restrict__ A, int lda,
                    const __bf16* __restrict__ B, int ldb,
                    float* __restrict__ C, int ldc,
                    __bf16* __restrict__ Cbf,
                    __bf16* __restrict__ CbfT, int ldt,
                    const float* __restrict__ bias,
                    int M, int N, int K)
{
  const int tid  = threadIdx.x;
  const int wave = tid >> 5;
  const int lane = tid & 31;
  const int wm   = wave & 3;        // 4 waves along M
  const int wn   = wave >> 2;       // 2 waves along N
  const int m0   = blockIdx.x * 128 + wm * 32;
  const int n0   = blockIdx.y * 64  + wn * 32;
  const int lm   = lane & 15;       // row/col within 16
  const int kh   = lane >> 4;       // K half selector

  v8f acc[2][2] = {};

  for (int k0 = 0; k0 < K; k0 += 32) {
    v16bf a[2], b[2];
    // ---- A fragments (16x32 bf16 layout: lane holds contiguous K = kh*16+e)
    #pragma unroll
    for (int i = 0; i < 2; ++i) {
      const __bf16* ap = A + (size_t)(m0 + i * 16 + lm) * lda + k0 + kh * 16;
      v8bf_t lo = *(const v8bf_t*)ap;
      v8bf_t hi = *(const v8bf_t*)(ap + 8);
      a[i] = __builtin_shufflevector(lo, hi, 0, 1, 2, 3, 4, 5, 6, 7,
                                              8, 9, 10, 11, 12, 13, 14, 15);
    }
    // ---- B fragments (32x16: lane = column of D, contiguous K = kh*16+e)
    #pragma unroll
    for (int j = 0; j < 2; ++j)
      b[j] = *(const v16bf*)(B + (size_t)(n0 + j * 16 + lm) * ldb + k0 + kh * 16);

    #pragma unroll
    for (int i = 0; i < 2; ++i)
      #pragma unroll
      for (int j = 0; j < 2; ++j)
        acc[i][j] = __builtin_amdgcn_wmma_f32_16x16x32_bf16(
            false, a[i], false, b[j], (short)0, acc[i][j], false, false);
  }

  // ---- store (C/D layout: lane col = lm, rows r + 8*kh)
  #pragma unroll
  for (int i = 0; i < 2; ++i) {
    #pragma unroll
    for (int j = 0; j < 2; ++j) {
      const int n = n0 + j * 16 + lm;
      const float bb = bias ? bias[n] : 0.0f;
      #pragma unroll
      for (int r = 0; r < 8; ++r) {
        const int m = m0 + i * 16 + r + kh * 8;
        const float v = acc[i][j][r] + bb;
        if (C)    C[(size_t)m * ldc + n]    = v;
        if (Cbf)  Cbf[(size_t)m * ldc + n]  = (__bf16)v;
        if (CbfT) CbfT[(size_t)n * ldt + m] = (__bf16)v;
      }
    }
  }
}

// ---------------------------------------------------------------------------
// Combined bias: outB[o] = sum_m W[o,m]*innerB[m] + outerB[o]   (o < 512)
// ---------------------------------------------------------------------------
__global__ __launch_bounds__(256)
void bias_comb_kernel(const float* __restrict__ W, const float* __restrict__ innerB,
                      const float* __restrict__ outerB, float* __restrict__ outB)
{
  const int o = blockIdx.x * 256 + threadIdx.x;
  if (o >= kE) return;
  float s = outerB[o];
  const float* wp = W + (size_t)o * kE;
  for (int m = 0; m < kE; ++m) s += wp[m] * innerB[m];
  outB[o] = s;
}

// d[b,h] = qq[b, h*64 : h*64+64] . bk'[h*64 : ...]
__global__ __launch_bounds__(256)
void dvec_kernel(const float* __restrict__ qq, const float* __restrict__ bkp,
                 float* __restrict__ d)
{
  const int idx = blockIdx.x * 256 + threadIdx.x;   // B*H
  const int b = idx >> 3, h = idx & 7;
  float s = 0.f;
  const float* qp = qq + (size_t)b * kE + h * 64;
  const float* bp = bkp + h * 64;
  #pragma unroll 8
  for (int e = 0; e < 64; ++e) s += qp[e] * bp[e];
  d[idx] = s;
}

// ---------------------------------------------------------------------------
// Per-graph logits: logits[n,h] = (x_n . c[b,h,:] + d[b,h]) * 1/sqrt(D)
// one workgroup per graph; wave-per-node; c[b] staged in LDS (16KB);
// X streamed with non-temporal loads + prefetch.
// ---------------------------------------------------------------------------
__global__ __launch_bounds__(256)
void logits_kernel(const float* __restrict__ X, const float* __restrict__ c,
                   const float* __restrict__ dvec, const int* __restrict__ batch,
                   float* __restrict__ logits)
{
  const int b = blockIdx.x;
  __shared__ float cs[kH * kE];
  __shared__ int seg[2];
  const int tid = threadIdx.x;
  for (int i = tid; i < kH * kE; i += 256) cs[i] = c[(size_t)b * (kH * kE) + i];
  if (tid == 0) { seg[0] = lowb(batch, kN, b); seg[1] = lowb(batch, kN, b + 1); }
  __syncthreads();
  const int start = seg[0], end = seg[1];
  const int wave = tid >> 5, lane = tid & 31;
  const float dv = (lane < kH) ? dvec[b * kH + lane] : 0.f;

  for (int n = start + wave; n < end; n += 8) {
    const f4v* xp = (const f4v*)(X + (size_t)n * kE + lane * 16);
    __builtin_prefetch(X + (size_t)(n + 8) * kE + lane * 16, 0, 1);
    f4v xv[4];
    #pragma unroll
    for (int q = 0; q < 4; ++q) xv[q] = __builtin_nontemporal_load(xp + q);

    float s[kH];
    #pragma unroll
    for (int h = 0; h < kH; ++h) {
      const f4v* cp = (const f4v*)(cs + h * kE + lane * 16);
      float t = 0.f;
      #pragma unroll
      for (int q = 0; q < 4; ++q) {
        f4v cv = cp[q];
        t += xv[q].x * cv.x + xv[q].y * cv.y + xv[q].z * cv.z + xv[q].w * cv.w;
      }
      s[h] = t;
    }
    #pragma unroll
    for (int h = 0; h < kH; ++h) {
      s[h] += __shfl_xor(s[h], 1, 32);
      s[h] += __shfl_xor(s[h], 2, 32);
      s[h] += __shfl_xor(s[h], 4, 32);
      s[h] += __shfl_xor(s[h], 8, 32);
      s[h] += __shfl_xor(s[h], 16, 32);
    }
    if (lane < kH) {
      float v = s[0];
      #pragma unroll
      for (int h = 1; h < kH; ++h) if (lane == h) v = s[h];
      logits[(size_t)n * kH + lane] = (v + dv) * 0.125f;   // 1/sqrt(64)
    }
  }
}

// ---------------------------------------------------------------------------
// Per-graph softmax + weighted segment sum of x:
//   y[b,h,:] = sum_{n in seg b} softmax(logits)[n,h] * x_n     (emitted as bf16)
// Deterministic: per-wave register accumulators merged serially via barriers.
// ---------------------------------------------------------------------------
__global__ __launch_bounds__(256)
void attn_y_kernel(const float* __restrict__ X, const float* __restrict__ lg,
                   const int* __restrict__ batch, __bf16* __restrict__ y)
{
  const int b = blockIdx.x;
  __shared__ float ylds[kH * kE];   // 16 KB
  __shared__ float ms[kH], zs[kH];
  __shared__ int seg[2];
  const int tid = threadIdx.x, wave = tid >> 5, lane = tid & 31;
  if (tid == 0) { seg[0] = lowb(batch, kN, b); seg[1] = lowb(batch, kN, b + 1); }
  for (int i = tid; i < kH * kE; i += 256) ylds[i] = 0.f;
  __syncthreads();
  const int start = seg[0], end = seg[1];

  // phase 1: wave == head -> segment max and normalizer
  {
    const int h = wave;
    float m = -3.4e38f;
    for (int n = start + lane; n < end; n += 32) m = fmaxf(m, lg[(size_t)n * kH + h]);
    #pragma unroll
    for (int o = 16; o >= 1; o >>= 1) m = fmaxf(m, __shfl_xor(m, o, 32));
    float Z = 0.f;
    for (int n = start + lane; n < end; n += 32) Z += __expf(lg[(size_t)n * kH + h] - m);
    #pragma unroll
    for (int o = 16; o >= 1; o >>= 1) Z += __shfl_xor(Z, o, 32);
    if (lane == 0) { ms[h] = m; zs[h] = Z; }
  }
  __syncthreads();
  float mh[kH], rz[kH];
  #pragma unroll
  for (int h = 0; h < kH; ++h) { mh[h] = ms[h]; rz[h] = 1.0f / zs[h]; }

  // phase 2: wave-per-node, register accumulation fy[h][16]
  float fy[kH][16];
  #pragma unroll
  for (int h = 0; h < kH; ++h)
    #pragma unroll
    for (int e = 0; e < 16; ++e) fy[h][e] = 0.f;

  for (int n = start + wave; n < end; n += 8) {
    const f4v* xp = (const f4v*)(X + (size_t)n * kE + lane * 16);
    __builtin_prefetch(X + (size_t)(n + 8) * kE + lane * 16, 0, 1);
    float xv[16];
    #pragma unroll
    for (int q = 0; q < 4; ++q) {
      f4v v = __builtin_nontemporal_load(xp + q);
      xv[q * 4 + 0] = v.x; xv[q * 4 + 1] = v.y; xv[q * 4 + 2] = v.z; xv[q * 4 + 3] = v.w;
    }
    const float* lp = lg + (size_t)n * kH;
    #pragma unroll
    for (int h = 0; h < kH; ++h) {
      const float w = __expf(lp[h] - mh[h]) * rz[h];
      #pragma unroll
      for (int e = 0; e < 16; ++e) fy[h][e] += w * xv[e];
    }
  }

  // deterministic serialized merge into LDS
  for (int w = 0; w < 8; ++w) {
    if (wave == w) {
      #pragma unroll
      for (int h = 0; h < kH; ++h)
        #pragma unroll
        for (int e = 0; e < 16; ++e)
          ylds[h * kE + lane * 16 + e] += fy[h][e];
    }
    __syncthreads();
  }
  for (int i = tid; i < kH * kE; i += 256)
    y[(size_t)b * (kH * kE) + i] = (__bf16)ylds[i];
}

// ---------------------------------------------------------------------------
// LayerNorm over last dim (512); wave per row, 8 rows per block
// ---------------------------------------------------------------------------
__global__ __launch_bounds__(256)
void ln_kernel(const float* __restrict__ in, const float* __restrict__ g,
               const float* __restrict__ be, float* __restrict__ out)
{
  const int row  = blockIdx.x * 8 + (threadIdx.x >> 5);
  const int lane = threadIdx.x & 31;
  const float* p = in + (size_t)row * kE + lane * 16;
  float xv[16];
  float s = 0.f, s2 = 0.f;
  #pragma unroll
  for (int e = 0; e < 16; ++e) { xv[e] = p[e]; s += xv[e]; s2 += xv[e] * xv[e]; }
  #pragma unroll
  for (int o = 16; o >= 1; o >>= 1) { s += __shfl_xor(s, o, 32); s2 += __shfl_xor(s2, o, 32); }
  const float mu  = s * (1.0f / kE);
  const float var = s2 * (1.0f / kE) - mu * mu;
  const float inv = rsqrtf(var + 1e-5f);
  float* op = out + (size_t)row * kE + lane * 16;
  const float* gp = g + lane * 16;
  const float* bp = be + lane * 16;
  #pragma unroll
  for (int e = 0; e < 16; ++e) op[e] = (xv[e] - mu) * inv * gp[e] + bp[e];
}

// ---------------------------------------------------------------------------
extern "C" void kernel_launch(void* const* d_in, const int* in_sizes, int n_in,
                              void* d_out, int out_size, void* d_ws, size_t ws_size,
                              hipStream_t stream) {
  const float* X    = (const float*)d_in[0];   // node_features [N,E]
  const float* S    = (const float*)d_in[1];   // scaffold_embedding [B,E]
  const float* Wq   = (const float*)d_in[2];
  const float* bq   = (const float*)d_in[3];
  const float* Wk   = (const float*)d_in[4];
  const float* bk   = (const float*)d_in[5];
  const float* Wv   = (const float*)d_in[6];
  const float* bv   = (const float*)d_in[7];
  const float* ipw  = (const float*)d_in[8];   // [3E,E]
  const float* ipb  = (const float*)d_in[9];   // [3E]
  const float* mow  = (const float*)d_in[10];
  const float* mob  = (const float*)d_in[11];
  const float* ow   = (const float*)d_in[12];
  const float* ob   = (const float*)d_in[13];
  const float* lng  = (const float*)d_in[14];
  const float* lnb  = (const float*)d_in[15];
  const int*   bat  = (const int*)d_in[16];
  float* outp = (float*)d_out;

  // ---- workspace carve-up ----
  char* p = (char*)d_ws;
  auto allocF  = [&](size_t n) { float*  r = (float*)p;  p += n * sizeof(float);  return r; };
  auto allocBF = [&](size_t n) { __bf16* r = (__bf16*)p; p += n * sizeof(__bf16); return r; };

  const size_t EE = (size_t)kE * kE;
  const size_t BE = (size_t)kB * kE;

  float* qq   = allocF(BE);                     // fp32 qq (for dvec)
  float* bkp  = allocF(kE);
  float* bvp  = allocF(kE);
  float* dvec = allocF((size_t)kB * kH);
  float* cbuf = allocF((size_t)kB * kH * kE);   // 64 MB
  float* lgb  = allocF((size_t)kN * kH);        // 4 MB
  float* o2   = allocF(BE);

  __bf16* Sbf    = allocBF(BE);
  __bf16* Wq_bf  = allocBF(EE);
  __bf16* ipw_bf = allocBF(3 * EE);
  __bf16* WkT_bf = allocBF(EE);   // Wk transposed
  __bf16* WvT_bf = allocBF(EE);   // Wv transposed
  __bf16* mow_bf = allocBF(EE);
  __bf16* ow_bf  = allocBF(EE);
  __bf16* q_bf   = allocBF(BE);
  __bf16* qq_bf  = allocBF(BE);
  __bf16* WkpT_bf = allocBF(EE);  // (Wik@Wk)^T, bf16
  __bf16* Wvp_bf  = allocBF(EE);  // (Wiv@Wv),  bf16
  __bf16* y_bf   = allocBF((size_t)kB * kH * kE);  // 32 MB
  __bf16* pl_bf  = allocBF(BE);
  __bf16* o1_bf  = allocBF(BE);

  const dim3 blk(256);
  auto cvtGrid = [](size_t n) { return dim3((unsigned)((n / 2 + 255) / 256)); };

  // 0) one-time bf16 conversions / transposes of GEMM inputs
  f32_to_bf16_kernel<<<cvtGrid(BE),     blk, 0, stream>>>(S,   Sbf,    (int)BE);
  f32_to_bf16_kernel<<<cvtGrid(EE),     blk, 0, stream>>>(Wq,  Wq_bf,  (int)EE);
  f32_to_bf16_kernel<<<cvtGrid(3 * EE), blk, 0, stream>>>(ipw, ipw_bf, (int)(3 * EE));
  f32_to_bf16_kernel<<<cvtGrid(EE),     blk, 0, stream>>>(mow, mow_bf, (int)EE);
  f32_to_bf16_kernel<<<cvtGrid(EE),     blk, 0, stream>>>(ow,  ow_bf,  (int)EE);
  transpose_f32_to_bf16<<<dim3(16, 16), blk, 0, stream>>>(Wk, WkT_bf, kE, kE);
  transpose_f32_to_bf16<<<dim3(16, 16), blk, 0, stream>>>(Wv, WvT_bf, kE, kE);

  // 1) q = S @ Wq.T + bq                         (bf16 out only)
  gemm_bf16_wmma<<<dim3(kB / 128, kE / 64), blk, 0, stream>>>(
      Sbf, kE, Wq_bf, kE, nullptr, kE, q_bf, nullptr, 0, bq, kB, kE, kE);
  // 2) qq = q @ Wiq.T + biq                      (fp32 for dvec + bf16)
  gemm_bf16_wmma<<<dim3(kB / 128, kE / 64), blk, 0, stream>>>(
      q_bf, kE, ipw_bf, kE, qq, kE, qq_bf, nullptr, 0, ipb, kB, kE, kE);
  // 3) Wk' = Wik @ Wk  -> emit (Wk')^T bf16 ;  Wv' = Wiv @ Wv -> bf16
  gemm_bf16_wmma<<<dim3(kE / 128, kE / 64), blk, 0, stream>>>(
      ipw_bf + EE, kE, WkT_bf, kE, nullptr, kE, nullptr, WkpT_bf, kE, nullptr, kE, kE, kE);
  gemm_bf16_wmma<<<dim3(kE / 128, kE / 64), blk, 0, stream>>>(
      ipw_bf + 2 * EE, kE, WvT_bf, kE, nullptr, kE, Wvp_bf, nullptr, 0, nullptr, kE, kE, kE);
  // 4) combined biases bk' = Wik@bk + bik ; bv' = Wiv@bv + biv
  bias_comb_kernel<<<dim3(2), blk, 0, stream>>>(ipw + EE, bk, ipb + kE, bkp);
  bias_comb_kernel<<<dim3(2), blk, 0, stream>>>(ipw + 2 * EE, bv, ipb + 2 * kE, bvp);
  // 5) d[b,h] = qq_head . bk'_head
  dvec_kernel<<<dim3((kB * kH) / 256), blk, 0, stream>>>(qq, bkp, dvec);
  // 6) per-head c[b,h,:] = qq[b,h,:] @ Wk'_head     (B operand = (Wk')^T slice)
  for (int h = 0; h < kH; ++h) {
    gemm_bf16_wmma<<<dim3(kB / 128, kE / 64), blk, 0, stream>>>(
        qq_bf + h * 64, kE, WkpT_bf + h * 64, kE,
        cbuf + (size_t)h * kE, kH * kE, nullptr, nullptr, 0, nullptr, kB, kE, 64);
  }
  // 7) logits pass over node_features
  logits_kernel<<<dim3(kB), blk, 0, stream>>>(X, cbuf, dvec, bat, lgb);
  // 8) softmax + weighted segment sums y (bf16)
  attn_y_kernel<<<dim3(kB), blk, 0, stream>>>(X, lgb, bat, y_bf);
  // 9) per-head pooled = y_head @ Wv'_head.T + bv'_head  (M=4096,N=64,K=512)
  for (int h = 0; h < kH; ++h) {
    gemm_bf16_wmma<<<dim3(kB / 128, 1), blk, 0, stream>>>(
        y_bf + (size_t)h * kE, kH * kE, Wvp_bf + (size_t)h * 64 * kE, kE,
        nullptr, kE, pl_bf + h * 64, nullptr, 0, bvp + h * 64, kB, 64, kE);
  }
  // 10) out projections
  gemm_bf16_wmma<<<dim3(kB / 128, kE / 64), blk, 0, stream>>>(
      pl_bf, kE, mow_bf, kE, nullptr, kE, o1_bf, nullptr, 0, mob, kB, kE, kE);
  gemm_bf16_wmma<<<dim3(kB / 128, kE / 64), blk, 0, stream>>>(
      o1_bf, kE, ow_bf, kE, o2, kE, nullptr, nullptr, 0, ob, kB, kE, kE);
  // 11) LayerNorm -> d_out
  ln_kernel<<<dim3(kB / 8), blk, 0, stream>>>(o2, lng, lnb, outp);
}
